// RNN_L_19782619366146
// MI455X (gfx1250) — matrix operand order repaired and checked by reading, not compiled, not run on hardware
//
#include <hip/hip_runtime.h>
#include <stdint.h>

// ---------------------------------------------------------------------------
// Types for CDNA5 WMMA (wave32): v_wmma_f32_16x16x32_bf16
// ---------------------------------------------------------------------------
typedef __bf16 bf16_t;
typedef bf16_t v16bf __attribute__((ext_vector_type(16)));
typedef float  v8f   __attribute__((ext_vector_type(8)));

struct FragB {
    union {
        v16bf v;
        uint4 q[2];   // 32 bytes = 16 x bf16
    };
};

__device__ __forceinline__ unsigned short f2bf_bits(float f) {
    bf16_t b = (bf16_t)f;                 // hardware RNE f32->bf16
    union { bf16_t b; unsigned short u; } cv;
    cv.b = b;
    return cv.u;
}

__device__ __forceinline__ float sigm(float x) {
    return 1.0f / (1.0f + __expf(-x));
}

__device__ __forceinline__ v8f wmma_bf16(const FragB& a, const FragB& b, v8f c) {
    // D = A(16x32 bf16) x B(32x16 bf16) + C(16x16 f32)
    return __builtin_amdgcn_wmma_f32_16x16x32_bf16(
        /*neg_a=*/false, a.v, /*neg_b=*/false, b.v,
        /*c_mod=*/(short)0, c, /*reuse_a=*/false, /*reuse_b=*/false);
}

// B fragment fetch from LDS (pre-swizzled, 32B contiguous per lane)
__device__ __forceinline__ FragB loadB(const uint4* lds, int nt, int kc, int lane) {
    FragB b;
    const uint4* p = lds + (((nt * 6 + kc) * 32 + lane) << 1);
    b.q[0] = p[0];
    b.q[1] = p[1];
    return b;
}

// A fragment from bf16 hidden-state row in LDS.
// ISA A-layout (16-bit 16x32): lane half=0: e0..7 <-> K 0..7,  e8..15 <-> K 16..23
//                              lane half=1: e0..7 <-> K 8..15, e8..15 <-> K 24..31
__device__ __forceinline__ FragB loadA_h(const bf16_t* row, int base, int hf) {
    FragB a;
    a.q[0] = *reinterpret_cast<const uint4*>(row + base + 8 * hf);
    a.q[1] = *reinterpret_cast<const uint4*>(row + base + 16 + 8 * hf);
    return a;
}

// A fragment from fp32 x in global memory, converted to bf16.
__device__ __forceinline__ FragB loadA_x(const float* xr, int hf) {
    float4 f0 = *reinterpret_cast<const float4*>(xr + 8 * hf);
    float4 f1 = *reinterpret_cast<const float4*>(xr + 8 * hf + 4);
    float4 f2 = *reinterpret_cast<const float4*>(xr + 16 + 8 * hf);
    float4 f3 = *reinterpret_cast<const float4*>(xr + 16 + 8 * hf + 4);
    FragB a;
    a.v[0]  = (bf16_t)f0.x; a.v[1]  = (bf16_t)f0.y; a.v[2]  = (bf16_t)f0.z; a.v[3]  = (bf16_t)f0.w;
    a.v[4]  = (bf16_t)f1.x; a.v[5]  = (bf16_t)f1.y; a.v[6]  = (bf16_t)f1.z; a.v[7]  = (bf16_t)f1.w;
    a.v[8]  = (bf16_t)f2.x; a.v[9]  = (bf16_t)f2.y; a.v[10] = (bf16_t)f2.z; a.v[11] = (bf16_t)f2.w;
    a.v[12] = (bf16_t)f3.x; a.v[13] = (bf16_t)f3.y; a.v[14] = (bf16_t)f3.z; a.v[15] = (bf16_t)f3.w;
    return a;
}

// ---------------------------------------------------------------------------
// Pack weights into per-lane B-fragment order (bf16), combined [x|h] K axis.
// W1c: 512 x 192 (K<64: w_ih1, else w_hh1).  W2c: 256 x 192 (K<128: w_ih2).
// Flat layout: idx = ((nt*6 + kc)*32 + lane)*16 + e
// B layout:    N = nt*16 + (lane&15),  K = kc*32 + (e&7) + 8*(lane>>4) + 16*(e>>3)
// ---------------------------------------------------------------------------
__global__ void pack_weights_kernel(const float* __restrict__ w_ih1,
                                    const float* __restrict__ w_hh1,
                                    const float* __restrict__ w_ih2,
                                    const float* __restrict__ w_hh2,
                                    unsigned short* __restrict__ w1p,
                                    unsigned short* __restrict__ w2p) {
    const int N1 = 32 * 6 * 32 * 16;  // 98304 elements (layer 1)
    const int N2 = 16 * 6 * 32 * 16;  // 49152 elements (layer 2)
    int idx = blockIdx.x * blockDim.x + threadIdx.x;
    if (idx < N1) {
        int e  = idx & 15;
        int l  = (idx >> 4) & 31;
        int tk = idx >> 9;           // nt*6 + kc
        int kc = tk % 6;
        int nt = tk / 6;
        int n  = nt * 16 + (l & 15);
        int K  = kc * 32 + (e & 7) + 8 * (l >> 4) + 16 * (e >> 3);
        float v = (K < 64) ? w_ih1[n * 64 + K] : w_hh1[n * 128 + (K - 64)];
        w1p[idx] = f2bf_bits(v);
    } else if (idx < N1 + N2) {
        int j  = idx - N1;
        int e  = j & 15;
        int l  = (j >> 4) & 31;
        int tk = j >> 9;
        int kc = tk % 6;
        int nt = tk / 6;
        int n  = nt * 16 + (l & 15);
        int K  = kc * 32 + (e & 7) + 8 * (l >> 4) + 16 * (e >> 3);
        float v = (K < 128) ? w_ih2[n * 128 + K] : w_hh2[n * 64 + (K - 128)];
        w2p[j] = f2bf_bits(v);
    }
}

// ---------------------------------------------------------------------------
// Fused 2-layer LSTM scan. One workgroup owns 16 batch rows for all 2048
// steps (batch rows are independent -> no inter-WG sync). 8 waves.
// Wave w owns hidden units j0=16w: tiles {w, w+8, w+16, w+24} give it
// i/f/g/o for the SAME hidden units -> gate math is thread-local.
// ---------------------------------------------------------------------------
__global__ __launch_bounds__(256, 1)
void lstm_scan_kernel(const float* __restrict__ x,
                      const float* __restrict__ b_ih1, const float* __restrict__ b_hh1,
                      const float* __restrict__ b_ih2, const float* __restrict__ b_hh2,
                      const float* __restrict__ fc1_w, const float* __restrict__ fc1_b,
                      const float* __restrict__ fc2_w, const float* __restrict__ fc2_b,
                      const uint4* __restrict__ w1p,   const uint4* __restrict__ w2p,
                      float* __restrict__ out) {
    __shared__ uint4 ldsW1[32 * 6 * 32 * 2];   // 196608 B
    __shared__ uint4 ldsW2[16 * 6 * 32 * 2];   //  98304 B
    __shared__ __align__(16) bf16_t hs1[16 * 128];  // 4 KB
    __shared__ __align__(16) bf16_t hs2[16 * 64];   // 2 KB
    __shared__ float bsum1[512];
    __shared__ float bsum2[256];
    __shared__ float h2f[16 * 64];
    __shared__ float vfc[64];
    __shared__ float biasc;

    const int tid  = threadIdx.x;
    const int wave = tid >> 5;
    const int lane = tid & 31;
    const int hf   = lane >> 4;     // which K-half of the wave
    const int col  = lane & 15;     // N column (and A-row m) for this lane
    const int b0   = blockIdx.x * 16;

    // ---- one-time staging into LDS ----
    for (int i = tid; i < 32 * 6 * 32 * 2; i += 256) ldsW1[i] = w1p[i];
    for (int i = tid; i < 16 * 6 * 32 * 2; i += 256) ldsW2[i] = w2p[i];
    for (int i = tid; i < 512; i += 256) bsum1[i] = b_ih1[i] + b_hh1[i];
    if (tid < 256) bsum2[tid] = b_ih2[tid] + b_hh2[tid];
    for (int i = tid; i < 16 * 128; i += 256) hs1[i] = (bf16_t)0.0f;
    for (int i = tid; i < 16 * 64;  i += 256) hs2[i] = (bf16_t)0.0f;
    if (tid < 64) {   // collapse fc2 o fc1 (both linear): v[j] = sum_k fc2[k]*fc1[k][j]
        float s = 0.0f;
        for (int k = 0; k < 32; ++k) s += fc2_w[k] * fc1_w[k * 64 + tid];
        vfc[tid] = s;
    }
    if (tid == 0) {
        float s = 0.0f;
        for (int k = 0; k < 32; ++k) s += fc2_w[k] * fc1_b[k];
        biasc = s + fc2_b[0];
    }
    __syncthreads();

    // per-thread persistent cell state (C/D layout: reg r <-> row r+8*hf, col jj)
    float c1r[8], c2r[8];
#pragma unroll
    for (int r = 0; r < 8; ++r) { c1r[r] = 0.0f; c2r[r] = 0.0f; }

    const int   jj1 = wave * 16 + col;         // layer-1 hidden index (0..127)
    const float bI = bsum1[jj1];
    const float bF = bsum1[128 + jj1];
    const float bG = bsum1[256 + jj1];
    const float bO = bsum1[384 + jj1];
    int jj2 = 0;
    float b2I = 0.f, b2F = 0.f, b2G = 0.f, b2O = 0.f;
    if (wave < 4) {
        jj2 = wave * 16 + col;                 // layer-2 hidden index (0..63)
        b2I = bsum2[jj2];
        b2F = bsum2[64 + jj2];
        b2G = bsum2[128 + jj2];
        b2O = bsum2[192 + jj2];
    }

    const float* xbase = x + (size_t)(b0 + col) * 2048 * 64;
    const v8f vzero = {0.f, 0.f, 0.f, 0.f, 0.f, 0.f, 0.f, 0.f};

    for (int t = 0; t < 2048; ++t) {
        // ---- phase 1: layer-1 gates [16,512] = [x_t | h1] @ W1c^T ----
        v8f aI = vzero, aF = vzero, aG = vzero, aO = vzero;
        const float* xt = xbase + (size_t)t * 64;
#pragma unroll
        for (int kc = 0; kc < 6; ++kc) {
            FragB a;
            if (kc < 2) a = loadA_x(xt + kc * 32, hf);                 // K 0..63  : x_t
            else        a = loadA_h(hs1 + col * 128, (kc - 2) * 32, hf); // K 64..191: h1
            aI = wmma_bf16(a, loadB(ldsW1, wave,      kc, lane), aI);
            aF = wmma_bf16(a, loadB(ldsW1, 8  + wave, kc, lane), aF);
            aG = wmma_bf16(a, loadB(ldsW1, 16 + wave, kc, lane), aG);
            aO = wmma_bf16(a, loadB(ldsW1, 24 + wave, kc, lane), aO);
        }
        __syncthreads();

        // ---- phase 2: layer-1 pointwise, write h1_t (bf16) ----
#pragma unroll
        for (int r = 0; r < 8; ++r) {
            int m = r + 8 * hf;
            float iv = sigm(aI[r] + bI);
            float fv = sigm(aF[r] + bF);
            float gv = tanhf(aG[r] + bG);
            float ov = sigm(aO[r] + bO);
            float c  = fv * c1r[r] + iv * gv;
            c1r[r]   = c;
            hs1[m * 128 + jj1] = (bf16_t)(ov * tanhf(c));
        }
        __syncthreads();

        // ---- phase 3: layer-2 gates [16,256] = [h1_t | h2] @ W2c^T (waves 0-3) ----
        v8f dI = vzero, dF = vzero, dG = vzero, dO = vzero;
        if (wave < 4) {
#pragma unroll
            for (int kc = 0; kc < 6; ++kc) {
                FragB a;
                if (kc < 4) a = loadA_h(hs1 + col * 128, kc * 32, hf);       // K 0..127 : h1_t
                else        a = loadA_h(hs2 + col * 64, (kc - 4) * 32, hf);  // K 128..191: h2
                dI = wmma_bf16(a, loadB(ldsW2, wave,      kc, lane), dI);
                dF = wmma_bf16(a, loadB(ldsW2, 4  + wave, kc, lane), dF);
                dG = wmma_bf16(a, loadB(ldsW2, 8  + wave, kc, lane), dG);
                dO = wmma_bf16(a, loadB(ldsW2, 12 + wave, kc, lane), dO);
            }
        }
        __syncthreads();

        // ---- phase 4: layer-2 pointwise, write h2_t (bf16; f32 on last step) ----
        if (wave < 4) {
#pragma unroll
            for (int r = 0; r < 8; ++r) {
                int m = r + 8 * hf;
                float iv = sigm(dI[r] + b2I);
                float fv = sigm(dF[r] + b2F);
                float gv = tanhf(dG[r] + b2G);
                float ov = sigm(dO[r] + b2O);
                float c  = fv * c2r[r] + iv * gv;
                c2r[r]   = c;
                float h  = ov * tanhf(c);
                hs2[m * 64 + jj2] = (bf16_t)h;
                if (t == 2047) h2f[m * 64 + jj2] = h;
            }
        }
        __syncthreads();
    }

    // ---- collapsed FC head: out[b] = h2_T . v + biasc ----
    if (tid < 16) {
        float acc = biasc;
#pragma unroll 8
        for (int j = 0; j < 64; ++j) acc += h2f[tid * 64 + j] * vfc[j];
        out[b0 + tid] = acc;
    }
}

// ---------------------------------------------------------------------------
extern "C" void kernel_launch(void* const* d_in, const int* in_sizes, int n_in,
                              void* d_out, int out_size, void* d_ws, size_t ws_size,
                              hipStream_t stream) {
    (void)in_sizes; (void)n_in; (void)out_size; (void)ws_size;
    const float* x     = (const float*)d_in[0];
    const float* w_ih1 = (const float*)d_in[1];
    const float* w_hh1 = (const float*)d_in[2];
    const float* b_ih1 = (const float*)d_in[3];
    const float* b_hh1 = (const float*)d_in[4];
    const float* w_ih2 = (const float*)d_in[5];
    const float* w_hh2 = (const float*)d_in[6];
    const float* b_ih2 = (const float*)d_in[7];
    const float* b_hh2 = (const float*)d_in[8];
    const float* fc1_w = (const float*)d_in[9];
    const float* fc1_b = (const float*)d_in[10];
    const float* fc2_w = (const float*)d_in[11];
    const float* fc2_b = (const float*)d_in[12];

    unsigned short* w1p = (unsigned short*)d_ws;          // 98304 bf16 = 196608 B
    unsigned short* w2p = w1p + 32 * 6 * 32 * 16;         // 49152 bf16 =  98304 B

    // 147456 total packed elements -> 576 blocks x 256 threads
    pack_weights_kernel<<<576, 256, 0, stream>>>(w_ih1, w_hh1, w_ih2, w_hh2, w1p, w2p);

    lstm_scan_kernel<<<16, 256, 0, stream>>>(x, b_ih1, b_hh1, b_ih2, b_hh2,
                                             fc1_w, fc1_b, fc2_w, fc2_b,
                                             (const uint4*)w1p, (const uint4*)w2p,
                                             (float*)d_out);
}
